// SpectralMamba_15092515078211
// MI455X (gfx1250) — compile-verified, matrix operand order checked by reading, not run on hardware
//
#include <hip/hip_runtime.h>
#include <hip/hip_bf16.h>

typedef __attribute__((ext_vector_type(16))) _Float16 v16h;
typedef __attribute__((ext_vector_type(8)))  _Float16 v8h;
typedef __attribute__((ext_vector_type(8)))  float    v8f;

#define D_MODEL   16
#define D_INNER   32
#define HW        16384      // 128*128 pixels
#define PIX_PER_BLOCK 16

__device__ __forceinline__ int kmap(int v, int h, int hi) {
  // 16-bit A/B operand K index for element (vgpr v, half h) on lane-half hi
  return ((v < 4) ? (2*v + h) : (16 + 2*(v-4) + h)) + hi*8;
}

__device__ __forceinline__ v8f wmma16(v16h a, v16h b, v8f c) {
  return __builtin_amdgcn_wmma_f32_16x16x32_f16(false, a, false, b, (short)0, c,
                                                false, false);
}

__device__ __forceinline__ float sigmoidf_(float x) { return 1.0f / (1.0f + __expf(-x)); }
__device__ __forceinline__ float softplusf_(float x) {
  return (x > 20.f) ? x : __logf(1.0f + __expf(x));
}

union H16 { v16h v; _Float16 h[16]; };

// ---------------------------------------------------------------------------
// Kernel 1: fused in_proj(WMMA) -> conv+SiLU -> x_proj(WMMA) -> scan ->
//           gate -> out_proj(WMMA), writes pre-GN activations + group sums.
// Block = 256 thr (8 waves), block handles 16 consecutive pixels, wave -> 2.
// ---------------------------------------------------------------------------
__global__ __launch_bounds__(256)
void mamba_kernel(const float* __restrict__ x,
                  const float* __restrict__ in_proj_w,
                  const float* __restrict__ conv_w,
                  const float* __restrict__ conv_b,
                  const float* __restrict__ x_proj_w,
                  const float* __restrict__ dt_proj_w,
                  const float* __restrict__ dt_proj_b,
                  const float* __restrict__ A_log,
                  const float* __restrict__ Dvec,
                  const float* __restrict__ out_proj_w,
                  float* __restrict__ mamba_out,
                  float* __restrict__ gsums)
{
  __shared__ __align__(16) float    s_io[128 * PIX_PER_BLOCK];   // 8KB in/out tile
  __shared__ __align__(16) float    s_dbl[8][16 * 48];           // 24KB (dt,B,C) rows
  __shared__ __align__(16) _Float16 s_xy[8][16 * 32];            // 8KB xs/y staging
  __shared__ float s_gs[4], s_gq[4];

  const int tid  = threadIdx.x;
  const int w    = tid >> 5;
  const int lane = tid & 31;
  const int hi   = (lane >> 4) & 1;   // which pixel of the wave's pair
  const int nlo  = lane & 15;
  const int p0   = blockIdx.x * PIX_PER_BLOCK;

  if (tid < 4) { s_gs[tid] = 0.f; s_gq[tid] = 0.f; }

  // ---- weight fragments (B operands, f32 -> f16), built once ----
  v16h Bin[4], Bxp[3], Bout;
  #pragma unroll
  for (int tile = 0; tile < 4; ++tile) {
    H16 u;
    #pragma unroll
    for (int e = 0; e < 16; ++e) {
      int k = kmap(e >> 1, e & 1, hi);
      u.h[e] = (_Float16)((k < D_MODEL) ? in_proj_w[(tile*16 + nlo)*D_MODEL + k] : 0.f);
    }
    Bin[tile] = u.v;
  }
  #pragma unroll
  for (int tile = 0; tile < 3; ++tile) {
    H16 u;
    #pragma unroll
    for (int e = 0; e < 16; ++e) {
      int k = kmap(e >> 1, e & 1, hi);
      int r = tile*16 + nlo;
      u.h[e] = (_Float16)((r < 33) ? x_proj_w[r*D_INNER + k] : 0.f);
    }
    Bxp[tile] = u.v;
  }
  {
    H16 u;
    #pragma unroll
    for (int e = 0; e < 16; ++e) {
      int k = kmap(e >> 1, e & 1, hi);
      u.h[e] = (_Float16)out_proj_w[nlo*D_INNER + k];
    }
    Bout = u.v;
  }

  // ---- per-lane channel params: this lane owns channels d0 and d0+16 ----
  const int d0 = nlo, d1 = nlo + 16;
  float cw0[4], cw1[4];
  #pragma unroll
  for (int k = 0; k < 4; ++k) { cw0[k] = conv_w[d0*4+k]; cw1[k] = conv_w[d1*4+k]; }
  const float cb0 = conv_b[d0],  cb1 = conv_b[d1];
  const float dtw0 = dt_proj_w[d0], dtw1 = dt_proj_w[d1];
  const float dtb0 = dt_proj_b[d0], dtb1 = dt_proj_b[d1];
  const float Dv0 = Dvec[d0], Dv1 = Dvec[d1];
  float Am0[16], Am1[16];
  #pragma unroll
  for (int s = 0; s < 16; ++s) {
    Am0[s] = -__expf(A_log[d0*16 + s]);
    Am1[s] = -__expf(A_log[d1*16 + s]);
  }

  // ---- stage input tile (128 ch x 16 px), coalesced ----
  #pragma unroll
  for (int it = 0; it < 8; ++it) {
    int c  = it*16 + (tid >> 4);
    int px = tid & 15;
    s_io[c*16 + px] = x[c*HW + p0 + px];
  }
  __syncthreads();

  // ---- A fragment for in_proj: row m = token (8*pix + l), K = d_model ----
  v16h a_in;
  {
    H16 u;
    const int m = nlo, pxl = 2*w + (m >> 3), ltk = m & 7;
    #pragma unroll
    for (int e = 0; e < 16; ++e) {
      int k = kmap(e >> 1, e & 1, hi);
      u.h[e] = (_Float16)((k < D_MODEL) ? s_io[(ltk*16 + k)*16 + pxl] : 0.f);
    }
    a_in = u.v;
  }

  const v8f zero = {0.f,0.f,0.f,0.f,0.f,0.f,0.f,0.f};
  v8f xi0 = wmma16(a_in, Bin[0], zero);   // xi channels 0-15
  v8f xi1 = wmma16(a_in, Bin[1], zero);   // xi channels 16-31
  v8f z0  = wmma16(a_in, Bin[2], zero);   // z  channels 0-15
  v8f z1  = wmma16(a_in, Bin[3], zero);   // z  channels 16-31

  // ---- depthwise causal conv (along VGPR index = token) + SiLU ----
  float xs0[8], xs1[8];
  #pragma unroll
  for (int l = 0; l < 8; ++l) {
    float a0 = cb0, a1 = cb1;
    #pragma unroll
    for (int k = 0; k < 4; ++k) {
      int j = l - 3 + k;
      if (j >= 0) { a0 += cw0[k]*xi0[j]; a1 += cw1[k]*xi1[j]; }
    }
    xs0[l] = a0 * sigmoidf_(a0);
    xs1[l] = a1 * sigmoidf_(a1);
  }

  // stage xs (f16) for x_proj A operand
  #pragma unroll
  for (int l = 0; l < 8; ++l) {
    int t = l + 8*hi;
    s_xy[w][t*32 + d0] = (_Float16)xs0[l];
    s_xy[w][t*32 + d1] = (_Float16)xs1[l];
  }
  __syncthreads();

  v16h a_xs;
  {
    v8h lo  = *(const v8h*)&s_xy[w][nlo*32 + hi*8];
    v8h hi8 = *(const v8h*)&s_xy[w][nlo*32 + 16 + hi*8];
    a_xs = __builtin_shufflevector(lo, hi8, 0,1,2,3,4,5,6,7,8,9,10,11,12,13,14,15);
  }
  v8f db0 = wmma16(a_xs, Bxp[0], zero);   // r = 0..15  (dt, B[0..14])
  v8f db1 = wmma16(a_xs, Bxp[1], zero);   // r = 16..31 (B[15], C[0..14])
  v8f db2 = wmma16(a_xs, Bxp[2], zero);   // r = 32     (C[15])

  #pragma unroll
  for (int v = 0; v < 8; ++v) {
    int t = v + 8*hi;
    s_dbl[w][t*48 + 0  + nlo] = db0[v];
    s_dbl[w][t*48 + 16 + nlo] = db1[v];
    s_dbl[w][t*48 + 32 + nlo] = db2[v];
  }
  __syncthreads();

  // ---- selective scan: per lane 2 channels x 16 states, f32 ----
  float h0[16], h1[16];
  #pragma unroll
  for (int s = 0; s < 16; ++s) { h0[s] = 0.f; h1[s] = 0.f; }

  #pragma unroll
  for (int l = 0; l < 8; ++l) {
    const int t = 8*hi + l;
    const float* row = &s_dbl[w][t*48];
    float dtr = row[0];
    float dt0 = softplusf_(dtr*dtw0 + dtb0);
    float dt1 = softplusf_(dtr*dtw1 + dtb1);
    float xv0 = xs0[l], xv1 = xs1[l];
    float y0 = 0.f, y1 = 0.f;
    #pragma unroll
    for (int s = 0; s < 16; ++s) {
      float Bm = row[1 + s];
      float Cm = row[17 + s];
      float dA0 = __expf(dt0 * Am0[s]);
      float dA1 = __expf(dt1 * Am1[s]);
      h0[s] = dA0*h0[s] + dt0*Bm*xv0;
      h1[s] = dA1*h1[s] + dt1*Bm*xv1;
      y0 += h0[s]*Cm;
      y1 += h1[s]*Cm;
    }
    float zz0 = z0[l], zz1 = z1[l];
    y0 = (y0 + xv0*Dv0) * (zz0 * sigmoidf_(zz0));
    y1 = (y1 + xv1*Dv1) * (zz1 * sigmoidf_(zz1));
    s_xy[w][t*32 + d0] = (_Float16)y0;
    s_xy[w][t*32 + d1] = (_Float16)y1;
  }
  __syncthreads();

  // ---- out_proj ----
  v16h a_y;
  {
    v8h lo  = *(const v8h*)&s_xy[w][nlo*32 + hi*8];
    v8h hi8 = *(const v8h*)&s_xy[w][nlo*32 + 16 + hi*8];
    a_y = __builtin_shufflevector(lo, hi8, 0,1,2,3,4,5,6,7,8,9,10,11,12,13,14,15);
  }
  v8f o = wmma16(a_y, Bout, zero);

  __syncthreads();                // safe to reuse s_io as output tile
  #pragma unroll
  for (int v = 0; v < 8; ++v) {
    int c = v*16 + nlo;           // output channel = token*16 + d_model idx
    s_io[c*16 + (2*w + hi)] = o[v];
  }
  __syncthreads();

  // ---- coalesced writeout + per-group sum/sumsq reduction ----
  float gs[4] = {0,0,0,0}, gq[4] = {0,0,0,0};
  #pragma unroll
  for (int it = 0; it < 8; ++it) {
    int c  = it*16 + (tid >> 4);
    int px = tid & 15;
    float val = s_io[c*16 + px];
    mamba_out[c*HW + p0 + px] = val;
    int g = c >> 5;
    gs[g] += val; gq[g] += val*val;
  }
  #pragma unroll
  for (int g = 0; g < 4; ++g) { atomicAdd(&s_gs[g], gs[g]); atomicAdd(&s_gq[g], gq[g]); }
  __syncthreads();
  if (tid < 4) { atomicAdd(&gsums[tid], s_gs[tid]); atomicAdd(&gsums[4+tid], s_gq[tid]); }
}

// ---------------------------------------------------------------------------
// Kernel 2: GroupNorm (from global sums) + SiLU + residual, vectorized x4.
// ---------------------------------------------------------------------------
__global__ __launch_bounds__(256)
void finalize_kernel(const float* __restrict__ x,
                     const float* __restrict__ mamba,
                     const float* __restrict__ sums,
                     const float* __restrict__ gamma,
                     const float* __restrict__ beta,
                     float* __restrict__ out)
{
  int i4  = blockIdx.x * blockDim.x + threadIdx.x;   // 0..524287
  int idx = i4 * 4;
  int c = idx >> 14;     // / 16384
  int g = c >> 5;
  const float inv = 1.0f / (32.0f * 16384.0f);
  float mean = sums[g] * inv;
  float var  = sums[4+g] * inv - mean*mean;
  float rstd = rsqrtf(var + 1e-5f);
  float ga = gamma[c], be = beta[c];
  float4 m4 = ((const float4*)mamba)[i4];
  float4 x4 = ((const float4*)x)[i4];
  float mv[4] = {m4.x, m4.y, m4.z, m4.w};
  float xv[4] = {x4.x, x4.y, x4.z, x4.w};
  float rv[4];
  #pragma unroll
  for (int j = 0; j < 4; ++j) {
    float gn = (mv[j] - mean) * rstd * ga + be;
    rv[j] = xv[j] + gn * sigmoidf_(gn);
  }
  float4 r; r.x = rv[0]; r.y = rv[1]; r.z = rv[2]; r.w = rv[3];
  ((float4*)out)[i4] = r;
}

__global__ void init_sums(float* g) { if (threadIdx.x < 16) g[threadIdx.x] = 0.f; }

extern "C" void kernel_launch(void* const* d_in, const int* in_sizes, int n_in,
                              void* d_out, int out_size, void* d_ws, size_t ws_size,
                              hipStream_t stream) {
  const float* x         = (const float*)d_in[0];
  const float* in_proj_w = (const float*)d_in[1];
  const float* conv_w    = (const float*)d_in[2];
  const float* conv_b    = (const float*)d_in[3];
  const float* x_proj_w  = (const float*)d_in[4];
  const float* dt_proj_w = (const float*)d_in[5];
  const float* dt_proj_b = (const float*)d_in[6];
  const float* A_log     = (const float*)d_in[7];
  const float* Dvec      = (const float*)d_in[8];
  const float* out_projw = (const float*)d_in[9];
  const float* gn_gamma  = (const float*)d_in[10];
  const float* gn_beta   = (const float*)d_in[11];

  float* ws    = (float*)d_ws;
  float* sums  = ws;          // 8 floats (+ pad to 16 for alignment)
  float* mamba = ws + 16;     // 2M floats = 8 MB staging

  init_sums<<<1, 32, 0, stream>>>(sums);
  mamba_kernel<<<dim3(HW / PIX_PER_BLOCK), dim3(256), 0, stream>>>(
      x, in_proj_w, conv_w, conv_b, x_proj_w, dt_proj_w, dt_proj_b,
      A_log, Dvec, out_projw, mamba, sums);
  finalize_kernel<<<dim3((HW * 128 / 4) / 256), dim3(256), 0, stream>>>(
      x, mamba, sums, gn_gamma, gn_beta, (float*)d_out);
}
